// TCRPeptideBindingModel_51591147159659
// MI455X (gfx1250) — matrix operand order repaired and checked by reading, not compiled, not run on hardware
//
#include <hip/hip_runtime.h>
#include <hip/hip_bf16.h>
#include <math.h>

// ---------------------------------------------------------------------------
// Model constants (match reference)
// ---------------------------------------------------------------------------
static const int D   = 480;   // input feature dim
static const int HID = 128;   // hidden dim (= H*C)
static const int HH  = 4;     // heads
static const int CC  = 32;    // channels per head
static const int NL  = 3;     // GAT layers
static const int Bn  = 2048;  // batch (graphs)

#define WPB 8                 // waves per GEMM block (8 * 16 = 128 output cols)
#define MT  4                 // 16-row tiles per block (64 rows)

typedef __attribute__((ext_vector_type(16))) __bf16 v16bf;
typedef __attribute__((ext_vector_type(8)))  float  v8f;

union FragBF { unsigned u[8]; v16bf v; };

__device__ __forceinline__ unsigned short f2bf(float f) {
  unsigned u = __float_as_uint(f);
  unsigned rnd = 0x7FFFu + ((u >> 16) & 1u);   // round-to-nearest-even
  return (unsigned short)((u + rnd) >> 16);
}

// order-preserving float<->uint encoding for atomicMax on floats
__device__ __forceinline__ unsigned ordenc(float f) {
  unsigned u = __float_as_uint(f);
  return (u & 0x80000000u) ? ~u : (u | 0x80000000u);
}
__device__ __forceinline__ float orddec(unsigned u) {
  return __uint_as_float((u & 0x80000000u) ? (u ^ 0x80000000u) : ~u);
}

// ---------------------------------------------------------------------------
// bf16 WMMA GEMM:  Out[M,Ncol] = act( A[M,K] @ W[K,Ncol] + bias )
// Wt = weights pre-transposed to bf16: Wt[n][k] (K-pairs contiguous).
// Block = 8 waves; each wave owns a 16-col stripe and computes MT=4 16x16
// tiles per k-step with a single (reused) B fragment:
//   per k-step/wave: 2x global_load_b128 (B) + 4x (8 LDS pair loads + WMMA).
// A tile (64x32 f32) is staged to LDS as packed bf16 pairs, one float2
// global load + one b32 ds_store per thread per pair (no divergence).
// REQUIREMENTS: K multiple of 32, M multiple of 64 (all call sites satisfy:
// M in {131072, 24576, 2048}); Ncol may be < 16*WPB (col-guarded stores).
// ---------------------------------------------------------------------------
__global__ __launch_bounds__(32 * WPB)
void gemm_bf16_wmma(const float* __restrict__ A,
                    const unsigned short* __restrict__ Wt,
                    const float* __restrict__ bias,
                    float* __restrict__ Out,
                    int M, int K, int Ncol, int relu)
{
  __shared__ unsigned Ls[16 * MT][17];  // 64 rows x 16 bf16-pairs (+1 pad)

  const int lane = threadIdx.x & 31;
  const int wave = threadIdx.x >> 5;
  const int m0   = blockIdx.x * (16 * MT);
  const int n0   = (blockIdx.y * WPB + wave) * 16;

  const int mrow = lane & 15;
  const int kbA  = (lane >> 4) * 16;   // A: lanes 0-15 -> K 0..15, lanes 16-31 -> 16..31
  const int kbB  = (lane >> 4) * 16;   // B: lanes 0-15 -> K 0..15, lanes 16-31 -> 16..31
  int ncl = n0 + (lane & 15);
  if (ncl > Ncol - 1) ncl = Ncol - 1;  // clamp (EXEC stays all-ones for WMMA)
  const unsigned short* __restrict__ wrow = Wt + (long long)ncl * K;

  v8f acc[MT];
#pragma unroll
  for (int rt = 0; rt < MT; ++rt) acc[rt] = (v8f){};

  for (int k0 = 0; k0 < K; k0 += 32) {
    // stage 64x32 A tile as packed bf16 pairs: 1024 pairs, 4 per thread
    // (M is a multiple of 64 -> no row guard, no divergence)
#pragma unroll
    for (int i = 0; i < 4; ++i) {
      int p   = threadIdx.x + i * 256;  // pair index
      int r   = p >> 4;                 // row in tile (16 pairs per row)
      int kp  = (p & 15) * 2;           // k offset of the pair
      float2 f = *(const float2*)&A[(long long)(m0 + r) * K + k0 + kp];
      Ls[r][p & 15] = (unsigned)f2bf(f.x) | ((unsigned)f2bf(f.y) << 16);
    }
    __syncthreads();

    FragBF b;
#pragma unroll
    for (int j = 0; j < 8; ++j)
      b.u[j] = *(const unsigned*)&wrow[k0 + kbB + 2 * j];

#pragma unroll
    for (int rt = 0; rt < MT; ++rt) {
      FragBF a;
#pragma unroll
      for (int j = 0; j < 8; ++j) {
        int kA = kbA + 2 * j;
        a.u[j] = Ls[rt * 16 + mrow][kA >> 1];
      }
      acc[rt] = __builtin_amdgcn_wmma_f32_16x16x32_bf16(
          /*neg_a=*/false, a.v, /*neg_b=*/false, b.v,
          /*c_mod=*/(short)0, acc[rt], /*reuse_a=*/false, /*reuse_b=*/false);
    }
    __syncthreads();
  }

  // C/D layout: VGPR i -> M = i + (lane>=16 ? 8 : 0), N = lane&15
  const int col = n0 + (lane & 15);
  const bool colok = (col < Ncol);
  float bv = 0.0f;
  if (bias && colok) bv = bias[col];
#pragma unroll
  for (int rt = 0; rt < MT; ++rt) {
#pragma unroll
    for (int i = 0; i < 8; ++i) {
      int row = m0 + rt * 16 + i + ((lane >> 4) * 8);
      if (colok) {
        float v = acc[rt][i] + bv;
        if (relu) v = fmaxf(v, 0.0f);
        Out[(long long)row * Ncol + col] = v;
      }
    }
  }
}

// ---------------------------------------------------------------------------
// Weight prep: W[K,Ncol] f32 -> Wt[Ncol,K] bf16 bits
// ---------------------------------------------------------------------------
__global__ void w_to_bf16t_kernel(const float* __restrict__ W,
                                  unsigned short* __restrict__ Wt,
                                  int K, int Ncol)
{
  long long i = (long long)blockIdx.x * blockDim.x + threadIdx.x;
  if (i >= (long long)K * Ncol) return;
  int k = (int)(i / Ncol), n = (int)(i % Ncol);
  Wt[(long long)n * K + k] = f2bf(W[i]);
}

// ---------------------------------------------------------------------------
// Fills
// ---------------------------------------------------------------------------
__global__ void fill_f32_kernel(float* p, float v, long long n) {
  long long i = (long long)blockIdx.x * blockDim.x + threadIdx.x;
  if (i < n) p[i] = v;
}
__global__ void fill_u32_kernel(unsigned* p, unsigned v, long long n) {
  long long i = (long long)blockIdx.x * blockDim.x + threadIdx.x;
  if (i < n) p[i] = v;
}

// ---------------------------------------------------------------------------
// Per-node attention scores: s_src[n,h] = sum_c hp[n,h,c]*a_src[h,c] (and dst)
// One node per 128-thread block; wave w == head w; wave32 shuffle reduction.
// ---------------------------------------------------------------------------
__global__ __launch_bounds__(128)
void scores_kernel(const float* __restrict__ hp,
                   const float* __restrict__ a_src,
                   const float* __restrict__ a_dst,
                   float* __restrict__ s_src, float* __restrict__ s_dst, int N)
{
  int n = blockIdx.x;
  int t = threadIdx.x;
  int w = t >> 5, c = t & 31;
  float hv = hp[(long long)n * HID + t];
  float ps = hv * a_src[w * CC + c];
  float pd = hv * a_dst[w * CC + c];
#pragma unroll
  for (int off = 16; off > 0; off >>= 1) {
    ps += __shfl_down(ps, off, 32);
    pd += __shfl_down(pd, off, 32);
  }
  if (c == 0) { s_src[n * HH + w] = ps; s_dst[n * HH + w] = pd; }
}

// ---------------------------------------------------------------------------
// Edge passes.  Edge id e in [0, E) is a real edge, e in [E, E+N) is the
// self-loop for node e-E (reference appends self loops for every node).
// ---------------------------------------------------------------------------
__device__ __forceinline__ void edge_nodes(const int* __restrict__ ei,
                                           long long e, int E, int& src, int& dst) {
  if (e < (long long)E) { src = ei[e]; dst = ei[(long long)E + e]; }
  else { src = dst = (int)(e - E); }
}

__global__ void edge_max_kernel(const int* __restrict__ ei, int E, int N,
                                const float* __restrict__ s_src,
                                const float* __restrict__ s_dst,
                                unsigned* __restrict__ m)
{
  long long idx = (long long)blockIdx.x * blockDim.x + threadIdx.x;
  long long tot = ((long long)E + N) * HH;
  if (idx >= tot) return;
  int h = (int)(idx & 3);
  long long e = idx >> 2;
  int src, dst; edge_nodes(ei, e, E, src, dst);
  float s = s_src[src * HH + h] + s_dst[dst * HH + h];
  float lre = s > 0.0f ? s : 0.2f * s;            // leaky_relu(0.2)
  atomicMax(&m[dst * HH + h], ordenc(lre));
}

__global__ void edge_sum_kernel(const int* __restrict__ ei, int E, int N,
                                const float* __restrict__ s_src,
                                const float* __restrict__ s_dst,
                                const unsigned* __restrict__ m,
                                float* __restrict__ z)
{
  long long idx = (long long)blockIdx.x * blockDim.x + threadIdx.x;
  long long tot = ((long long)E + N) * HH;
  if (idx >= tot) return;
  int h = (int)(idx & 3);
  long long e = idx >> 2;
  int src, dst; edge_nodes(ei, e, E, src, dst);
  float s = s_src[src * HH + h] + s_dst[dst * HH + h];
  float lre = s > 0.0f ? s : 0.2f * s;
  float p = __expf(lre - orddec(m[dst * HH + h]));
  atomicAdd(&z[dst * HH + h], p);
}

// one edge per 128-thread block: agg[dst,:] += alpha[h] * hp[src,:]
__global__ __launch_bounds__(128)
void edge_agg_kernel(const int* __restrict__ ei, int E, int N,
                     const float* __restrict__ s_src,
                     const float* __restrict__ s_dst,
                     const unsigned* __restrict__ m,
                     const float* __restrict__ z,
                     const float* __restrict__ hp,
                     float* __restrict__ agg)
{
  __shared__ float alpha_s[HH];
  long long e = blockIdx.x;
  int src, dst; edge_nodes(ei, e, E, src, dst);
  int t = threadIdx.x;
  if (t < HH) {
    int h = t;
    float s = s_src[src * HH + h] + s_dst[dst * HH + h];
    float lre = s > 0.0f ? s : 0.2f * s;
    float p = __expf(lre - orddec(m[dst * HH + h]));
    alpha_s[h] = p / z[dst * HH + h];
  }
  __syncthreads();
  float msg = hp[(long long)src * HID + t] * alpha_s[t >> 5];
  atomicAdd(&agg[(long long)dst * HID + t], msg);
}

// ---------------------------------------------------------------------------
// Post: h[n] = relu(layernorm(agg[n] + bg)) + h[n]      (one node per block)
// ---------------------------------------------------------------------------
__global__ __launch_bounds__(128)
void gat_post_kernel(const float* __restrict__ agg, const float* __restrict__ bg,
                     const float* __restrict__ g, const float* __restrict__ bta,
                     float* __restrict__ h, int N)
{
  __shared__ float red[128];
  int n = blockIdx.x, t = threadIdx.x;
  long long base = (long long)n * HID;
  float v = agg[base + t] + bg[t];
  red[t] = v; __syncthreads();
#pragma unroll
  for (int s = 64; s > 0; s >>= 1) { if (t < s) red[t] += red[t + s]; __syncthreads(); }
  float mu = red[0] * (1.0f / 128.0f);
  __syncthreads();
  float d = v - mu;
  red[t] = d * d; __syncthreads();
#pragma unroll
  for (int s = 64; s > 0; s >>= 1) { if (t < s) red[t] += red[t + s]; __syncthreads(); }
  float var = red[0] * (1.0f / 128.0f);
  float y = d * rsqrtf(var + 1e-5f) * g[t] + bta[t];
  y = fmaxf(y, 0.0f);
  h[base + t] = y + h[base + t];
}

// ---------------------------------------------------------------------------
// Pooling (batch ids are sorted, values < Bn)
// ---------------------------------------------------------------------------
__global__ void pool_add_kernel(const float* __restrict__ h, const int* __restrict__ batch,
                                float* __restrict__ pool, int N)
{
  long long idx = (long long)blockIdx.x * blockDim.x + threadIdx.x;
  if (idx >= (long long)N * HID) return;
  int n = (int)(idx >> 7), c = (int)(idx & 127);
  atomicAdd(&pool[(long long)batch[n] * HID + c], h[idx]);
}
__global__ void pool_cnt_kernel(const int* __restrict__ batch, float* __restrict__ cnt, int N)
{
  long long i = (long long)blockIdx.x * blockDim.x + threadIdx.x;
  if (i < N) atomicAdd(&cnt[batch[i]], 1.0f);
}
__global__ void pool_div_kernel(const float* __restrict__ pool, const float* __restrict__ cnt,
                                float* __restrict__ zcat, int zoff)
{
  long long idx = (long long)blockIdx.x * blockDim.x + threadIdx.x;
  if (idx >= (long long)Bn * HID) return;
  int b = (int)(idx >> 7), c = (int)(idx & 127);
  zcat[(long long)b * (2 * HID) + zoff + c] = pool[idx] / fmaxf(cnt[b], 1.0f);
}

// ---------------------------------------------------------------------------
// Final classifier layer: 64 -> 1 + sigmoid
// ---------------------------------------------------------------------------
__global__ void cls3_kernel(const float* __restrict__ z2, const float* __restrict__ Wc3,
                            const float* __restrict__ bc3, float* __restrict__ out, int B)
{
  int b = blockIdx.x * blockDim.x + threadIdx.x;
  if (b >= B) return;
  float acc = bc3[0];
#pragma unroll
  for (int k = 0; k < 64; ++k) acc += z2[b * 64 + k] * Wc3[k];
  out[b] = 1.0f / (1.0f + __expf(-acc));
}

// ---------------------------------------------------------------------------
// Host-side orchestration
// ---------------------------------------------------------------------------
static inline unsigned divup_ll(long long a, int b) { return (unsigned)((a + b - 1) / b); }

static void launch_gemm(const float* A, const unsigned short* Wt, const float* bias,
                        float* Out, int M, int K, int Ncol, int relu, hipStream_t s)
{
  dim3 grid((M + 16 * MT - 1) / (16 * MT), (Ncol + 16 * WPB - 1) / (16 * WPB));
  gemm_bf16_wmma<<<grid, 32 * WPB, 0, s>>>(A, Wt, bias, Out, M, K, Ncol, relu);
}

static void encode_graph(const float* x, const int* ei, const int* batch, int N, int E,
                         const float* b_in, const float* bg,
                         const float* att_src, const float* att_dst,
                         const float* ln_g, const float* ln_b,
                         const unsigned short* wint, const unsigned short* wgt,
                         float* h, float* hp, float* agg,
                         float* ssrc, float* sdst, unsigned* mbuf, float* zbuf,
                         float* pool, float* cnt, float* zcat, int zoff,
                         hipStream_t stream)
{
  // h0 = relu(x @ W_in + b_in)
  launch_gemm(x, wint, b_in, h, N, D, HID, 1, stream);

  long long Etot = (long long)E + N;
  for (int l = 0; l < NL; ++l) {
    launch_gemm(h, wgt + (size_t)l * HID * HID, nullptr, hp, N, HID, HID, 0, stream);
    scores_kernel<<<N, 128, 0, stream>>>(hp, att_src + l * HH * CC, att_dst + l * HH * CC,
                                         ssrc, sdst, N);
    fill_u32_kernel<<<divup_ll((long long)N * HH, 256), 256, 0, stream>>>(mbuf, 0u, (long long)N * HH);
    fill_f32_kernel<<<divup_ll((long long)N * HH, 256), 256, 0, stream>>>(zbuf, 0.0f, (long long)N * HH);
    fill_f32_kernel<<<divup_ll((long long)N * HID, 256), 256, 0, stream>>>(agg, 0.0f, (long long)N * HID);
    long long t4 = Etot * HH;
    edge_max_kernel<<<divup_ll(t4, 256), 256, 0, stream>>>(ei, E, N, ssrc, sdst, mbuf);
    edge_sum_kernel<<<divup_ll(t4, 256), 256, 0, stream>>>(ei, E, N, ssrc, sdst, mbuf, zbuf);
    edge_agg_kernel<<<(unsigned)Etot, 128, 0, stream>>>(ei, E, N, ssrc, sdst, mbuf, zbuf, hp, agg);
    gat_post_kernel<<<N, 128, 0, stream>>>(agg, bg + l * HID, ln_g + l * HID, ln_b + l * HID, h, N);
  }

  fill_f32_kernel<<<divup_ll((long long)Bn * HID, 256), 256, 0, stream>>>(pool, 0.0f, (long long)Bn * HID);
  fill_f32_kernel<<<divup_ll(Bn, 256), 256, 0, stream>>>(cnt, 0.0f, Bn);
  pool_add_kernel<<<divup_ll((long long)N * HID, 256), 256, 0, stream>>>(h, batch, pool, N);
  pool_cnt_kernel<<<divup_ll(N, 256), 256, 0, stream>>>(batch, cnt, N);
  pool_div_kernel<<<divup_ll((long long)Bn * HID, 256), 256, 0, stream>>>(pool, cnt, zcat, zoff);
}

extern "C" void kernel_launch(void* const* d_in, const int* in_sizes, int n_in,
                              void* d_out, int out_size, void* d_ws, size_t ws_size,
                              hipStream_t stream)
{
  (void)n_in; (void)out_size; (void)ws_size;
  const float* tcr_x     = (const float*)d_in[0];
  const int*   tcr_ei    = (const int*)  d_in[1];
  const int*   tcr_batch = (const int*)  d_in[2];
  const float* pep_x     = (const float*)d_in[3];
  const int*   pep_ei    = (const int*)  d_in[4];
  const int*   pep_batch = (const int*)  d_in[5];
  const float* W_in    = (const float*)d_in[6];
  const float* b_in    = (const float*)d_in[7];
  const float* Wg      = (const float*)d_in[8];
  const float* bg      = (const float*)d_in[9];
  const float* att_src = (const float*)d_in[10];
  const float* att_dst = (const float*)d_in[11];
  const float* ln_g    = (const float*)d_in[12];
  const float* ln_b    = (const float*)d_in[13];
  const float* Wc1     = (const float*)d_in[14];
  const float* bc1     = (const float*)d_in[15];
  const float* Wc2     = (const float*)d_in[16];
  const float* bc2     = (const float*)d_in[17];
  const float* Wc3     = (const float*)d_in[18];
  const float* bc3     = (const float*)d_in[19];
  float* out = (float*)d_out;

  const int Nt = in_sizes[0] / D;
  const int Et = in_sizes[1] / 2;
  const int Np = in_sizes[3] / D;
  const int Ep = in_sizes[4] / 2;
  const int Nmax = Nt > Np ? Nt : Np;

  // --- workspace layout (256B aligned) ---
  char* ws = (char*)d_ws;
  size_t off = 0;
  auto walloc = [&](size_t bytes) -> char* {
    char* p = ws + off;
    off = (off + bytes + 255) & ~(size_t)255;
    return p;
  };
  unsigned short* wint = (unsigned short*)walloc((size_t)HID * D * 2);          // [128][480]
  unsigned short* wgt  = (unsigned short*)walloc((size_t)NL * HID * HID * 2);   // 3x [128][128]
  unsigned short* wc1t = (unsigned short*)walloc((size_t)HID * (2 * HID) * 2);  // [128][256]
  unsigned short* wc2t = (unsigned short*)walloc((size_t)(HID / 2) * HID * 2);  // [64][128]
  float* zcat = (float*)walloc((size_t)Bn * 2 * HID * 4);
  float* pool = (float*)walloc((size_t)Bn * HID * 4);
  float* cnt  = (float*)walloc((size_t)Bn * 4);
  float* z1   = (float*)walloc((size_t)Bn * HID * 4);
  float* z2   = (float*)walloc((size_t)Bn * (HID / 2) * 4);
  float* hbuf = (float*)walloc((size_t)Nmax * HID * 4);
  float* hp   = (float*)walloc((size_t)Nmax * HID * 4);
  float* agg  = (float*)walloc((size_t)Nmax * HID * 4);
  float* ssrc = (float*)walloc((size_t)Nmax * HH * 4);
  float* sdst = (float*)walloc((size_t)Nmax * HH * 4);
  unsigned* mbuf = (unsigned*)walloc((size_t)Nmax * HH * 4);
  float* zbuf = (float*)walloc((size_t)Nmax * HH * 4);

  // --- weight prep: transpose + bf16 convert (cheap, deterministic each call)
  w_to_bf16t_kernel<<<divup_ll((long long)D * HID, 256), 256, 0, stream>>>(W_in, wint, D, HID);
  for (int l = 0; l < NL; ++l)
    w_to_bf16t_kernel<<<divup_ll((long long)HID * HID, 256), 256, 0, stream>>>(
        Wg + (size_t)l * HID * HID, wgt + (size_t)l * HID * HID, HID, HID);
  w_to_bf16t_kernel<<<divup_ll((long long)2 * HID * HID, 256), 256, 0, stream>>>(Wc1, wc1t, 2 * HID, HID);
  w_to_bf16t_kernel<<<divup_ll((long long)HID * (HID / 2), 256), 256, 0, stream>>>(Wc2, wc2t, HID, HID / 2);

  // --- encode both graphs (pep reuses the big buffers)
  encode_graph(tcr_x, tcr_ei, tcr_batch, Nt, Et, b_in, bg, att_src, att_dst, ln_g, ln_b,
               wint, wgt, hbuf, hp, agg, ssrc, sdst, mbuf, zbuf, pool, cnt, zcat, 0, stream);
  encode_graph(pep_x, pep_ei, pep_batch, Np, Ep, b_in, bg, att_src, att_dst, ln_g, ln_b,
               wint, wgt, hbuf, hp, agg, ssrc, sdst, mbuf, zbuf, pool, cnt, zcat, HID, stream);

  // --- classifier MLP (WMMA for the two big layers)
  launch_gemm(zcat, wc1t, bc1, z1, Bn, 2 * HID, HID, 1, stream);
  launch_gemm(z1,   wc2t, bc2, z2, Bn, HID, HID / 2, 1, stream);
  cls3_kernel<<<divup_ll(Bn, 256), 256, 0, stream>>>(z2, Wc3, bc3, out, Bn);
}
